// GAT_87436944212396
// MI455X (gfx1250) — matrix operand order repaired and checked by reading, not compiled, not run on hardware
//
#include <hip/hip_runtime.h>
#include <hip/hip_bf16.h>

// ---- problem constants (match reference) ----
#define HNUM 4
#define CNUM 64
#define HC   256   // H*C
#define GNUM 256
#define NEG_SLOPE 0.2f

typedef __attribute__((ext_vector_type(16))) _Float16 v16h;
typedef __attribute__((ext_vector_type(8)))  float    v8f;

// order-preserving float<->uint encoding for atomicMax on floats
__device__ __forceinline__ unsigned f32_ord(float f) {
  unsigned u = __float_as_uint(f);
  return (u & 0x80000000u) ? ~u : (u | 0x80000000u);
}
__device__ __forceinline__ float ord_f32(unsigned u) {
  return __uint_as_float((u & 0x80000000u) ? (u ^ 0x80000000u) : ~u);
}
__device__ __forceinline__ float lrelu(float z) {
  return z > 0.f ? z : NEG_SLOPE * z;
}

// ---------------------------------------------------------------------------
// Pack W [K, 256] (f32, row-major) into WMMA-B fragment order (f16):
// B layout used here (contiguous k; KxN 32x16 per k-step): lane l holds column
// n = l&15; lanes 0-15 hold K=0..15, lanes 16-31 hold K=16..31; element i
// within the 16-half fragment holds k_local = i + 16*(l>>4).
// Wp index: (((colTile * (K/32) + ks) * 32 + lane) * 16 + i)
// ---------------------------------------------------------------------------
__global__ void pack_w(const float* __restrict__ W, _Float16* __restrict__ Wp,
                       int K) {
  int t = blockIdx.x * blockDim.x + threadIdx.x;
  if (t >= K * HC) return;
  const int k = t / HC;
  const int n = t % HC;
  const int colTile = n >> 4;
  const int ks    = k >> 5;
  const int khalf = (k >> 4) & 1;
  const int i     = k & 15;
  const int lane  = (n & 15) + (khalf << 4);
  const int KS    = K >> 5;
  Wp[(((size_t)colTile * KS + ks) * 32 + lane) * 16 + i] = (_Float16)W[t];
}

// ---------------------------------------------------------------------------
// out[M,256] = act[M,K] @ W[K,256]   (f16 WMMA, f32 accumulate)
// block: 256 threads = 8 waves, one 16-row tile per block; wave w computes
// col tiles 2w and 2w+1 (two WMMAs sharing one A fragment).
// A tile is staged in LDS in fragment order:
//   A lane layout used here (contiguous k, 16x32): lane = {m = lane&15, khalf = lane>>4},
//   element i holds k_local = i + 16*khalf.
// D layout: VGPR r, lane l -> row r + 8*(l>=16), col l&15.
// ---------------------------------------------------------------------------
template <int K>
__global__ __launch_bounds__(256)
void gat_gemm_f16(const float* __restrict__ act,
                  const _Float16* __restrict__ Wp,
                  float* __restrict__ out, int M) {
  constexpr int KS = K / 32;
  __shared__ alignas(32) _Float16 lA[KS * 512];   // [ks][lane][16]

  const int t    = threadIdx.x;
  const int row0 = blockIdx.x * 16;

  // cooperative load + f32->f16 convert of the 16 x K activation tile,
  // written directly in WMMA-A fragment order
  for (int j = t; j < 16 * K; j += 256) {
    const int m  = j / K;
    const int k  = j % K;
    const int ks = k >> 5;
    const int kl = k & 31;
    const int khalf = (kl >> 4) & 1;
    const int i     = kl & 15;
    const int lane  = (m & 15) + (khalf << 4);
    lA[ks * 512 + lane * 16 + i] =
        (_Float16)act[(size_t)(row0 + m) * K + k];
  }
  __syncthreads();

  const int lane = t & 31;
  const int wave = t >> 5;

  const _Float16* bp0 = Wp + ((size_t)(2 * wave)     * KS * 32 + lane) * 16;
  const _Float16* bp1 = Wp + ((size_t)(2 * wave + 1) * KS * 32 + lane) * 16;

  union { v8f v; float e[8]; } C0, C1;
#pragma unroll
  for (int r = 0; r < 8; ++r) { C0.e[r] = 0.f; C1.e[r] = 0.f; }

#pragma unroll
  for (int ks = 0; ks < KS; ++ks) {
    const v16h av = *(const v16h*)(lA + ks * 512 + lane * 16); // 2x ds_load_b128
    const v16h b0 = *(const v16h*)(bp0 + ks * 512);            // 2x global_load_b128
    const v16h b1 = *(const v16h*)(bp1 + ks * 512);
    C0.v = __builtin_amdgcn_wmma_f32_16x16x32_f16(
        false, av, false, b0, (short)0, C0.v, false, false);
    C1.v = __builtin_amdgcn_wmma_f32_16x16x32_f16(
        false, av, false, b1, (short)0, C1.v, false, false);
  }

  const int n_lo = lane & 15;
#pragma unroll
  for (int r = 0; r < 8; ++r) {
    const int mm = r + ((lane >> 4) << 3);
    float* orow = out + (size_t)(row0 + mm) * HC + wave * 32;
    orow[n_lo]      = C0.e[r];
    orow[16 + n_lo] = C1.e[r];
  }
}

// alpha_s / alpha_d per (node, head): 64-wide dot products
__global__ void gat_alpha(const float* __restrict__ h,
                          const float* __restrict__ a_src,
                          const float* __restrict__ a_dst,
                          float* __restrict__ as_, float* __restrict__ ad_,
                          int NH) {
  int t = blockIdx.x * blockDim.x + threadIdx.x;
  if (t >= NH) return;
  int hh = t & (HNUM - 1);
  int nd = t >> 2;
  const float* hp = h + (size_t)nd * HC + hh * CNUM;
  const float* sp = a_src + hh * CNUM;
  const float* dp = a_dst + hh * CNUM;
  float ss = 0.f, dd = 0.f;
#pragma unroll 8
  for (int c = 0; c < CNUM; ++c) { float v = hp[c]; ss += v * sp[c]; dd += v * dp[c]; }
  as_[t] = ss;
  ad_[t] = dd;
}

// seed per-(dst,head) running max with the self-loop logit
__global__ void gat_max_init(const float* __restrict__ as_,
                             const float* __restrict__ ad_,
                             unsigned* __restrict__ maxb, int NH) {
  int t = blockIdx.x * blockDim.x + threadIdx.x;
  if (t >= NH) return;
  maxb[t] = f32_ord(lrelu(as_[t] + ad_[t]));
}

__global__ void gat_edge_max(const int* __restrict__ ei,
                             const float* __restrict__ as_,
                             const float* __restrict__ ad_,
                             unsigned* __restrict__ maxb, int E) {
  int t = blockIdx.x * blockDim.x + threadIdx.x;
  if (t >= E * HNUM) return;
  int hh = t & (HNUM - 1);
  int e  = t >> 2;
  int s  = ei[e];
  int d  = ei[E + e];
  float logit = lrelu(as_[s * HNUM + hh] + ad_[d * HNUM + hh]);
  atomicMax(&maxb[d * HNUM + hh], f32_ord(logit));
}

// exp-sum over incoming edges (edges [0,E) + self loops [E,E+N))
__global__ void gat_exp_sum(const int* __restrict__ ei,
                            const float* __restrict__ as_,
                            const float* __restrict__ ad_,
                            const unsigned* __restrict__ maxb,
                            float* __restrict__ sumb, int E, int N) {
  int t = blockIdx.x * blockDim.x + threadIdx.x;
  if (t >= (E + N) * HNUM) return;
  int hh = t & (HNUM - 1);
  int e  = t >> 2;
  int s, d;
  if (e < E) { s = ei[e]; d = ei[E + e]; } else { s = d = e - E; }
  float logit = lrelu(as_[s * HNUM + hh] + ad_[d * HNUM + hh]);
  float mx = ord_f32(maxb[d * HNUM + hh]);
  atomicAdd(&sumb[d * HNUM + hh], __expf(logit - mx));
}

// normalize + scatter message: acc[d,h,:] += alpha * h[s,h,:]
__global__ void gat_scatter(const int* __restrict__ ei,
                            const float* __restrict__ as_,
                            const float* __restrict__ ad_,
                            const unsigned* __restrict__ maxb,
                            const float* __restrict__ sumb,
                            const float* __restrict__ hbuf,
                            float* __restrict__ acc, int E, int N) {
  int t = blockIdx.x * blockDim.x + threadIdx.x;
  if (t >= (E + N) * HNUM) return;
  int hh = t & (HNUM - 1);
  int e  = t >> 2;
  int s, d;
  if (e < E) { s = ei[e]; d = ei[E + e]; } else { s = d = e - E; }
  float logit = lrelu(as_[s * HNUM + hh] + ad_[d * HNUM + hh]);
  float mx    = ord_f32(maxb[d * HNUM + hh]);
  float alpha = __expf(logit - mx) / sumb[d * HNUM + hh];
  const float* hs = hbuf + (size_t)s * HC + hh * CNUM;
  float*       op = acc  + (size_t)d * HC + hh * CNUM;
#pragma unroll
  for (int c = 0; c < CNUM; c += 4) {
    float4 v = *(const float4*)(hs + c);
    atomicAdd(op + c + 0, alpha * v.x);
    atomicAdd(op + c + 1, alpha * v.y);
    atomicAdd(op + c + 2, alpha * v.z);
    atomicAdd(op + c + 3, alpha * v.w);
  }
}

// head mean + bias + ReLU -> next-layer activation [N, C]
__global__ void gat_finalize(const float* __restrict__ acc,
                             const float* __restrict__ bias,
                             float* __restrict__ xout, int N) {
  int t = blockIdx.x * blockDim.x + threadIdx.x;
  if (t >= N * CNUM) return;
  int c  = t & (CNUM - 1);
  int nd = t >> 6;
  const float* ap = acc + (size_t)nd * HC;
  float m = 0.25f * (ap[c] + ap[CNUM + c] + ap[2 * CNUM + c] + ap[3 * CNUM + c]) + bias[c];
  xout[t] = fmaxf(m, 0.f);
}

// global mean pool accumulation
__global__ void gat_pool(const float* __restrict__ xin,
                         const int* __restrict__ batch,
                         float* __restrict__ pooled, float* __restrict__ cnt,
                         int N) {
  int t = blockIdx.x * blockDim.x + threadIdx.x;
  if (t >= N * CNUM) return;
  int c  = t & (CNUM - 1);
  int nd = t >> 6;
  int g  = batch[nd];
  atomicAdd(&pooled[g * CNUM + c], xin[t]);
  if (c == 0) atomicAdd(&cnt[g], 1.f);
}

// pooled/cnt @ lin_w + lin_b  -> out[G]
__global__ void gat_head(const float* __restrict__ pooled,
                         const float* __restrict__ cnt,
                         const float* __restrict__ lw,
                         const float* __restrict__ lb,
                         float* __restrict__ out) {
  int g = threadIdx.x;   // one block of G threads
  float inv = 1.f / fmaxf(cnt[g], 1.f);
  float s = 0.f;
#pragma unroll 8
  for (int c = 0; c < CNUM; ++c) s += pooled[g * CNUM + c] * inv * lw[c];
  out[g] = s + lb[0];
}

// ---------------------------------------------------------------------------
extern "C" void kernel_launch(void* const* d_in, const int* in_sizes, int n_in,
                              void* d_out, int out_size, void* d_ws, size_t ws_size,
                              hipStream_t stream) {
  const int N = in_sizes[0] / 128;     // 50000
  const int E = in_sizes[1] / 2;       // 400000
  const int IN = 128;

  const float* x     = (const float*)d_in[0];
  const int*   ei    = (const int*)d_in[1];
  const int*   batch = (const int*)d_in[2];
  const float* W[3]    = {(const float*)d_in[3], (const float*)d_in[7],  (const float*)d_in[11]};
  const float* asrc[3] = {(const float*)d_in[4], (const float*)d_in[8],  (const float*)d_in[12]};
  const float* adst[3] = {(const float*)d_in[5], (const float*)d_in[9],  (const float*)d_in[13]};
  const float* bias[3] = {(const float*)d_in[6], (const float*)d_in[10], (const float*)d_in[14]};
  const float* lin_w = (const float*)d_in[15];
  const float* lin_b = (const float*)d_in[16];
  float* out = (float*)d_out;

  // workspace carve-up (d_ws is 256B-aligned)
  _Float16* wpack = (_Float16*)d_ws;                     // [<=128*256] packed W (64KB)
  float*    hbuf  = (float*)((char*)d_ws + 65536);       // [N, 256]
  float*    accb  = hbuf + (size_t)N * HC;               // [N, 256]
  float*    xbuf  = accb + (size_t)N * HC;               // [N, 64]
  float*    as_   = xbuf + (size_t)N * CNUM;             // [N, 4]
  float*    ad_   = as_ + (size_t)N * HNUM;              // [N, 4]
  unsigned* maxb  = (unsigned*)(ad_ + (size_t)N * HNUM); // [N, 4]
  float*    sumb  = (float*)(maxb + (size_t)N * HNUM);   // [N, 4]
  float*    pooled = sumb + (size_t)N * HNUM;            // [G, 64]
  float*    cnt    = pooled + (size_t)GNUM * CNUM;       // [G]

  const int NH = N * HNUM;
  const int EH = E * HNUM;
  const int TH = (E + N) * HNUM;
  const int BLK = 256;

  for (int L = 0; L < 3; ++L) {
    const float* act  = (L == 0) ? x : xbuf;
    const int    Kdim = (L == 0) ? IN : CNUM;

    pack_w<<<(Kdim * HC + BLK - 1) / BLK, BLK, 0, stream>>>(W[L], wpack, Kdim);

    if (L == 0)
      gat_gemm_f16<128><<<N / 16, BLK, 0, stream>>>(act, wpack, hbuf, N);
    else
      gat_gemm_f16<64><<<N / 16, BLK, 0, stream>>>(act, wpack, hbuf, N);

    gat_alpha<<<(NH + BLK - 1) / BLK, BLK, 0, stream>>>(hbuf, asrc[L], adst[L], as_, ad_, NH);
    gat_max_init<<<(NH + BLK - 1) / BLK, BLK, 0, stream>>>(as_, ad_, maxb, NH);
    gat_edge_max<<<(EH + BLK - 1) / BLK, BLK, 0, stream>>>(ei, as_, ad_, maxb, E);

    hipMemsetAsync(sumb, 0, (size_t)NH * sizeof(float), stream);
    gat_exp_sum<<<(TH + BLK - 1) / BLK, BLK, 0, stream>>>(ei, as_, ad_, maxb, sumb, E, N);

    hipMemsetAsync(accb, 0, (size_t)N * HC * sizeof(float), stream);
    gat_scatter<<<(TH + BLK - 1) / BLK, BLK, 0, stream>>>(ei, as_, ad_, maxb, sumb, hbuf, accb, E, N);

    gat_finalize<<<(N * CNUM + BLK - 1) / BLK, BLK, 0, stream>>>(accb, bias[L], xbuf, N);
  }

  // global mean pool + linear head
  hipMemsetAsync(pooled, 0, (size_t)(GNUM * CNUM + GNUM) * sizeof(float), stream);
  gat_pool<<<(N * CNUM + BLK - 1) / BLK, BLK, 0, stream>>>(xbuf, batch, pooled, cnt, N);
  gat_head<<<1, GNUM, 0, stream>>>(pooled, cnt, lin_w, lin_b, out);
}
